// QuantLinear_65859028517116
// MI455X (gfx1250) — compile-verified, matrix operand order checked by reading, not run on hardware
//
#include <hip/hip_runtime.h>

typedef __bf16 bf16;
typedef bf16  bf16x8  __attribute__((ext_vector_type(8)));
typedef bf16  bf16x16 __attribute__((ext_vector_type(16)));
typedef float floatx8 __attribute__((ext_vector_type(8)));
typedef float floatx4 __attribute__((ext_vector_type(4)));

#define K_DIM 4096
#define N_DIM 11008
#define NP8   (N_DIM / 8)
#define MT    128
#define NT    128
#define KT    128              // == quant GROUP -> scale/zero constant per chunk
#define NC    (K_DIM / KT)     // 32 chunks
#define LDSK  136              // K + 8 pad (272B row stride, 16B aligned)

__global__ __launch_bounds__(256)
void w4a16_wmma_gemm(const float* __restrict__ x,
                     const int*   __restrict__ qweight,
                     const float* __restrict__ scales,
                     const int*   __restrict__ qzeros,
                     const float* __restrict__ bias,
                     float* __restrict__ out) {
  // double-buffered tiles: 2 * (34816 + 34816) = 139264 B of the 320 KB WGP pool
  __shared__ __align__(16) bf16 As[2][MT * LDSK];   // x tile  [m][k] bf16
  __shared__ __align__(16) bf16 Bs[2][NT * LDSK];   // W^T tile [n][k] bf16

  const int tid  = threadIdx.x;
  const int lane = tid & 31;          // wave32
  const int wave = tid >> 5;          // 0..7
  const int wm   = wave & 1;          // 2 waves over M (64 rows each)
  const int wn   = wave >> 1;         // 4 waves over N (32 cols each)

  const int bm = blockIdx.y * MT;
  const int bn = blockIdx.x * NT;

  // staging maps
  const int xrow  = tid >> 1;         // one x row per 2 threads
  const int xhalf = (tid & 1) * 64;   // 64 floats each
  const int nloc  = tid & 127;        // dequant: column within tile
  const int qhalf = (tid >> 7) * 8;   // 8 packed K-rows per thread
  const int ncol  = bn + nloc;

  const float* xbase = x + (size_t)(bm + xrow) * K_DIM + xhalf;

  // ---- register staging for the software pipeline ----
  floatx4  xf[16];     // raw fp32 x chunk (converted at store time)
  unsigned qr[8];      // packed 4-bit weights
  float    s;          // group scale
  int      zq;         // group zero (+1)

  auto load_stage = [&](int c) {
    const float* xp = xbase + c * KT;
    __builtin_prefetch(xp + KT, 0, 0);                 // warm L2 for chunk c+1
#pragma unroll
    for (int j = 0; j < 16; ++j)
      xf[j] = *(const floatx4*)(xp + j * 4);
    const int qrow0 = c * (KT / 8) + qhalf;
#pragma unroll
    for (int r = 0; r < 8; ++r)
      qr[r] = (unsigned)qweight[(size_t)(qrow0 + r) * N_DIM + ncol];
    s  = scales[(size_t)c * N_DIM + ncol];
    zq = (int)((qzeros[(size_t)c * NP8 + (ncol >> 3)] >> ((ncol & 7) * 4)) & 15) + 1;
  };

  auto store_stage = [&](int b) {
    bf16* Ab = &As[b][0];
    bf16* Bb = &Bs[b][0];
#pragma unroll
    for (int j = 0; j < 8; ++j) {
      floatx4 f0 = xf[2 * j], f1 = xf[2 * j + 1];
      bf16x8 h;
      h[0] = (bf16)f0[0]; h[1] = (bf16)f0[1]; h[2] = (bf16)f0[2]; h[3] = (bf16)f0[3];
      h[4] = (bf16)f1[0]; h[5] = (bf16)f1[1]; h[6] = (bf16)f1[2]; h[7] = (bf16)f1[3];
      *(bf16x8*)&Ab[xrow * LDSK + xhalf + j * 8] = h;
    }
#pragma unroll
    for (int r = 0; r < 8; ++r) {
      bf16x8 wv;
#pragma unroll
      for (int nib = 0; nib < 8; ++nib) {
        const int qv = (int)((qr[r] >> (nib * 4)) & 15u) - zq;
        wv[nib] = (bf16)((float)qv * s);
      }
      *(bf16x8*)&Bb[nloc * LDSK + (qhalf + r) * 8] = wv;
    }
  };

  floatx8 acc[4][2];
#pragma unroll
  for (int mt = 0; mt < 4; ++mt)
#pragma unroll
    for (int nt = 0; nt < 2; ++nt)
      acc[mt][nt] = (floatx8)0.0f;

  auto compute = [&](int b) {
    const bf16* Ab = &As[b][0];
    const bf16* Bb = &Bs[b][0];
#pragma unroll
    for (int ks = 0; ks < KT; ks += 32) {
      // 16-bit A layout: lanes 0-15 -> K {0..7,16..23}; lanes 16-31 -> K {8..15,24..31}
      const int klo = ks + ((lane < 16) ? 0 : 8);
      bf16x16 a[4], bfr[2];
#pragma unroll
      for (int mt = 0; mt < 4; ++mt) {
        const int mr = wm * 64 + mt * 16 + (lane & 15);
        union { bf16x16 v; bf16x8 h[2]; } u;
        u.h[0] = *(const bf16x8*)&Ab[mr * LDSK + klo];
        u.h[1] = *(const bf16x8*)&Ab[mr * LDSK + klo + 16];
        a[mt] = u.v;
      }
#pragma unroll
      for (int nt = 0; nt < 2; ++nt) {
        const int nr = wn * 32 + nt * 16 + (lane & 15);
        union { bf16x16 v; bf16x8 h[2]; } u;
        u.h[0] = *(const bf16x8*)&Bb[nr * LDSK + klo];
        u.h[1] = *(const bf16x8*)&Bb[nr * LDSK + klo + 16];
        bfr[nt] = u.v;
      }
#pragma unroll
      for (int mt = 0; mt < 4; ++mt)
#pragma unroll
        for (int nt = 0; nt < 2; ++nt)
          acc[mt][nt] = __builtin_amdgcn_wmma_f32_16x16x32_bf16(
              false, a[mt], false, bfr[nt], (short)0, acc[mt][nt], false, false);
    }
  };

  // ---- pipeline: prologue stages chunk 0 ----
  load_stage(0);
  store_stage(0);
  __syncthreads();

#pragma unroll 1
  for (int c = 0; c < NC; ++c) {
    if (c + 1 < NC) load_stage(c + 1);      // global loads issue before compute
    compute(c & 1);                          // 32 WMMAs hide the load latency
    if (c + 1 < NC) store_stage((c + 1) & 1);// cvt/dequant + ds_store to other buffer
    __syncthreads();                         // one barrier per chunk
  }

  // ---- epilogue: C layout (VGPR r -> M = r + 8*(lane>=16), N = lane%16) ----
#pragma unroll
  for (int nt = 0; nt < 2; ++nt) {
    const int col = bn + wn * 32 + nt * 16 + (lane & 15);
    const float bv = bias[col];
#pragma unroll
    for (int mt = 0; mt < 4; ++mt) {
      const int rbase = bm + wm * 64 + mt * 16 + ((lane >> 4) * 8);
#pragma unroll
      for (int r = 0; r < 8; ++r)
        out[(size_t)(rbase + r) * N_DIM + col] = acc[mt][nt][r] + bv;
    }
  }
}

extern "C" void kernel_launch(void* const* d_in, const int* in_sizes, int n_in,
                              void* d_out, int out_size, void* d_ws, size_t ws_size,
                              hipStream_t stream) {
  const float* x   = (const float*)d_in[0];
  const int*   qw  = (const int*)  d_in[1];
  const float* sc  = (const float*)d_in[2];
  const int*   qz  = (const int*)  d_in[3];
  const float* bia = (const float*)d_in[4];
  float*       out = (float*)d_out;

  const int M = in_sizes[0] / K_DIM;            // 8192 (4*2048)
  dim3 grid(N_DIM / NT, M / MT);                // 86 x 64 workgroups
  w4a16_wmma_gemm<<<grid, dim3(256), 0, stream>>>(x, qw, sc, qz, bia, out);
}
